// Qwen3VLVisionAttention_19138374271039
// MI455X (gfx1250) — matrix operand order repaired and checked by reading, not compiled, not run on hardware
//
#include <hip/hip_runtime.h>
#include <hip/hip_bf16.h>
#include <math.h>

typedef __attribute__((ext_vector_type(16))) _Float16 v16h;
typedef __attribute__((ext_vector_type(8)))  _Float16 v8h;
typedef __attribute__((ext_vector_type(8)))  float    v8f;
typedef int v4i_ __attribute__((vector_size(16)));   // matches builtin param pointee

#define TT 2048
#define DD 1280
#define NH 16
#define HD 80
#define HDP 96
#define NEGBIG (-3.0e38f)

static __device__ __forceinline__ v8f wmma32(v16h a, v16h b, v8f c) {
    // D = A(16x32 f16) * B(32x16 f16) + C(16x16 f32)
    return __builtin_amdgcn_wmma_f32_16x16x32_f16(false, a, false, b, (short)0, c, false, false);
}

// ---- gfx1250 async global->LDS copy (ASYNCcnt path), compile-guarded ----
#if __has_builtin(__builtin_amdgcn_global_load_async_to_lds_b128)
#define ASYNC_CP16(dst, src)                                                   \
    __builtin_amdgcn_global_load_async_to_lds_b128(                            \
        (__attribute__((address_space(1))) v4i_*)(src),                        \
        (__attribute__((address_space(3))) v4i_*)(dst), 0, 0)
#if __has_builtin(__builtin_amdgcn_s_wait_asynccnt)
#define ASYNC_JOIN() __builtin_amdgcn_s_wait_asynccnt(0)
#else
#define ASYNC_JOIN() asm volatile("s_wait_asynccnt 0" ::: "memory")
#endif
#else
#define ASYNC_CP16(dst, src) (*(v8h*)(dst) = *(const v8h*)(src))
#define ASYNC_JOIN() ((void)0)
#endif

// ---------------- f32 -> f16 convert ----------------
__global__ void k_cvt(const float* __restrict__ in, _Float16* __restrict__ out, int n) {
    int i = blockIdx.x * blockDim.x + threadIdx.x;
    if (i < n) out[i] = (_Float16)in[i];
}

// W[K x N] f32 -> Wt[N x K] f16 (so WMMA B-loads are contiguous per lane)
__global__ void k_cvt_t(const float* __restrict__ in, _Float16* __restrict__ out, int K, int N) {
    int i = blockIdx.x * blockDim.x + threadIdx.x;
    if (i < K * N) {
        int k = i / N, n = i - k * N;
        out[(size_t)n * K + k] = (_Float16)in[i];
    }
}

// ---------------- WMMA GEMM: C[MxN] = A16[MxK] * Bt16[NxK] + bias ----------------
// 256 threads = 8 waves; each wave a 32x64 register tile (2 M x 4 N WMMA tiles),
// block tile 64(M) x 256(N). 8 WMMAs per 12 b128 loads.
__global__ void __launch_bounds__(256) k_gemm(const _Float16* __restrict__ A,
                                              const _Float16* __restrict__ Bt,
                                              const float* __restrict__ bias,
                                              float* __restrict__ C,
                                              int M, int N, int K) {
    int lane = threadIdx.x & 31;
    int wid  = threadIdx.x >> 5;
    int m0 = blockIdx.y * 64  + (wid >> 2) * 32;
    int n0 = blockIdx.x * 256 + (wid & 3) * 64;
    int l15 = lane & 15;
    int hi  = (lane & 16) ? 1 : 0;
    int hi8 = hi << 3;

    const _Float16* ar0 = A + (size_t)(m0 + l15) * K + hi * 16;    // A layout: lanes>=16 hold K+16
    const _Float16* ar1 = ar0 + (size_t)16 * K;
    const _Float16* br0 = Bt + (size_t)(n0 + l15) * K + hi * 16;   // B layout: lanes>=16 hold K+16
    const _Float16* br1 = br0 + (size_t)16 * K;
    const _Float16* br2 = br0 + (size_t)32 * K;
    const _Float16* br3 = br0 + (size_t)48 * K;

    v8f acc[2][4] = {};
    for (int k0 = 0; k0 < K; k0 += 32) {
        v16h a0, a1;
        {
            v8h lo0 = *(const v8h*)(ar0 + k0), h0 = *(const v8h*)(ar0 + k0 + 8);
            v8h lo1 = *(const v8h*)(ar1 + k0), h1 = *(const v8h*)(ar1 + k0 + 8);
            #pragma unroll
            for (int i = 0; i < 8; i++) {
                a0[i] = lo0[i]; a0[i + 8] = h0[i];
                a1[i] = lo1[i]; a1[i + 8] = h1[i];
            }
        }
        v16h b0 = *(const v16h*)(br0 + k0);
        v16h b1 = *(const v16h*)(br1 + k0);
        v16h b2 = *(const v16h*)(br2 + k0);
        v16h b3 = *(const v16h*)(br3 + k0);
        acc[0][0] = wmma32(a0, b0, acc[0][0]); acc[0][1] = wmma32(a0, b1, acc[0][1]);
        acc[0][2] = wmma32(a0, b2, acc[0][2]); acc[0][3] = wmma32(a0, b3, acc[0][3]);
        acc[1][0] = wmma32(a1, b0, acc[1][0]); acc[1][1] = wmma32(a1, b1, acc[1][1]);
        acc[1][2] = wmma32(a1, b2, acc[1][2]); acc[1][3] = wmma32(a1, b3, acc[1][3]);
    }
    #pragma unroll
    for (int jt = 0; jt < 4; jt++) {
        float bv = bias[n0 + jt * 16 + l15];
        #pragma unroll
        for (int mt = 0; mt < 2; mt++)
            #pragma unroll
            for (int r = 0; r < 8; r++)
                C[(size_t)(m0 + mt * 16 + r + hi8) * N + n0 + jt * 16 + l15] = acc[mt][jt][r] + bv;
    }
}

// ---------------- RoPE + pack q/k (pad 80->96) and v (transposed) ----------------
__global__ void k_rope_pack(const float* __restrict__ qkv, const float* __restrict__ rope,
                            _Float16* __restrict__ qh, _Float16* __restrict__ kh,
                            _Float16* __restrict__ vt) {
    int b = blockIdx.x;
    int t = b / NH, n = b - t * NH;
    int h = threadIdx.x;                       // 0..95
    const float scale = 0.111803398874989485f; // 1/sqrt(80), folded into Q
    size_t qkbase = (size_t)t * (3 * DD) + (size_t)n * HD;
    size_t orow = ((size_t)n * TT + t) * HDP;
    if (h < HD) {
        float v = qkv[qkbase + 2 * DD + h];
        vt[((size_t)n * HD + h) * TT + t] = (_Float16)v;
        int hr = (h < 40) ? h : h - 40;
        float s, c;
        __sincosf(rope[t * 40 + hr], &s, &c);
        float q0 = qkv[qkbase + hr],      q1 = qkv[qkbase + hr + 40];
        float k0 = qkv[qkbase + DD + hr], k1 = qkv[qkbase + DD + hr + 40];
        float qo = (h < 40) ? (q0 * c - q1 * s) : (q0 * s + q1 * c);
        float ko = (h < 40) ? (k0 * c - k1 * s) : (k0 * s + k1 * c);
        qh[orow + h] = (_Float16)(qo * scale);
        kh[orow + h] = (_Float16)ko;
    } else if (h < HDP) {
        qh[orow + h] = (_Float16)0.0f;   // zero pad: contributes nothing to QK^T
        kh[orow + h] = (_Float16)0.0f;
    }
}

// ---------------- flash attention ----------------
// Block = 8 waves sharing one head + 8 consecutive 16-query blocks; K/V tiles for
// each 32-key step staged once per block into LDS (async-to-LDS when available).
// Scores computed as S^T = K_tile x Q^T so P lands lane-locally in WMMA A-layout.
__global__ void __launch_bounds__(256) k_attn(const _Float16* __restrict__ qh,
                                              const _Float16* __restrict__ kh,
                                              const _Float16* __restrict__ vt,
                                              const int* __restrict__ seg,
                                              _Float16* __restrict__ ao) {
    __shared__ int segs[TT];                                            // 8 KB
    __shared__ __attribute__((aligned(64))) _Float16 ksh[32 * HDP];     // 6 KB
    __shared__ __attribute__((aligned(64))) _Float16 vsh[HD * 32];      // 5 KB
    for (int i = threadIdx.x; i < TT; i += 256) segs[i] = seg[i];

    int lane = threadIdx.x & 31;
    int wid  = threadIdx.x >> 5;
    int head = blockIdx.x >> 4;
    int q0   = ((blockIdx.x & 15) * 8 + wid) * 16;
    int l15  = lane & 15;
    int hi   = (lane & 16) ? 1 : 0;
    int hi8  = hi << 3;

    // Preload B1 = Q^T (96 x 16): loop-invariant, 3 chunks of K=32
    const _Float16* qrow = qh + ((size_t)head * TT + q0 + l15) * HDP + hi * 16;
    v16h bq0 = *(const v16h*)(qrow + 0);
    v16h bq1 = *(const v16h*)(qrow + 32);
    v16h bq2 = *(const v16h*)(qrow + 64);

    const _Float16* khead = kh + (size_t)head * TT * HDP;
    const _Float16* vhead = vt + (size_t)head * HD * TT;

    __syncthreads();
    int sq = segs[q0 + l15];

    float m = NEGBIG, l = 0.0f;
    v8f a0 = {}, a1v = {}, a2v = {}, a3v = {}, a4v = {};

    for (int kb = 0; kb < TT; kb += 32) {
        // ---- cooperative stage: K tile 32x96 (384 16B chunks) + V tile 80x32 (320) ----
        __syncthreads();   // previous iteration fully consumed
        for (int c = threadIdx.x; c < 704; c += 256) {
            if (c < 384) {
                int row = c / 12, col = c - row * 12;     // 12 x 16B per 192B K row
                ASYNC_CP16(ksh + row * HDP + col * 8,
                           khead + (size_t)(kb + row) * HDP + col * 8);
            } else {
                int c2 = c - 384;
                int row = c2 >> 2, part = c2 & 3;          // 4 x 16B per 64B V row
                ASYNC_CP16(vsh + row * 32 + part * 8,
                           vhead + (size_t)row * TT + kb + ((part & 1) * 2 + (part >> 1)) * 8);
            }
        }
        ASYNC_JOIN();
        __syncthreads();

        // ---- scores S^T for 32 keys: two 16-key tiles, 3 WMMAs each (from LDS) ----
        v8f s0 = {}, s1 = {};
        const _Float16* k0row = ksh + l15 * HDP;
        const _Float16* k1row = ksh + (16 + l15) * HDP;
        #pragma unroll
        for (int c = 0; c < 3; c++) {
            v8h x0 = *(const v8h*)(k0row + c * 32 + hi * 16);
            v8h y0 = *(const v8h*)(k0row + c * 32 + 8 + hi * 16);
            v8h x1 = *(const v8h*)(k1row + c * 32 + hi * 16);
            v8h y1 = *(const v8h*)(k1row + c * 32 + 8 + hi * 16);
            v16h ka, kb2;
            #pragma unroll
            for (int i = 0; i < 8; i++) { ka[i] = x0[i];  ka[i + 8] = y0[i];
                                          kb2[i] = x1[i]; kb2[i + 8] = y1[i]; }
            v16h bqc = (c == 0) ? bq0 : (c == 1) ? bq1 : bq2;
            s0 = wmma32(ka,  bqc, s0);
            s1 = wmma32(kb2, bqc, s1);
        }
        // ---- segment mask + online softmax (per lane: one query column) ----
        float mloc = NEGBIG;
        #pragma unroll
        for (int r = 0; r < 8; r++) {
            s0[r] = (segs[kb + r + hi8]      == sq) ? s0[r] : -1.0e9f;
            s1[r] = (segs[kb + 16 + r + hi8] == sq) ? s1[r] : -1.0e9f;
            mloc = fmaxf(mloc, fmaxf(s0[r], s1[r]));
        }
        mloc = fmaxf(mloc, __shfl_xor(mloc, 16, 32));   // combine wave32 halves
        float mnew  = fmaxf(m, mloc);
        float alpha = __expf(m - mnew);
        float lloc = 0.0f;
        v16h p;   // P in WMMA A-layout, lane-local thanks to S^T orientation
        #pragma unroll
        for (int r = 0; r < 8; r++) {
            float p0 = __expf(s0[r] - mnew);
            float p1 = __expf(s1[r] - mnew);
            lloc += p0 + p1;
            p[r]     = (_Float16)p0;
            p[r + 8] = (_Float16)p1;
        }
        lloc += __shfl_xor(lloc, 16, 32);
        l = l * alpha + lloc;
        m = mnew;
        // per-output-row alpha via ds_bpermute (acc rows are r+hi8)
        float ar[8];
        #pragma unroll
        for (int r = 0; r < 8; r++) ar[r] = __shfl(alpha, r + hi8, 32);
        #pragma unroll
        for (int r = 0; r < 8; r++) {
            a0[r] *= ar[r]; a1v[r] *= ar[r]; a2v[r] *= ar[r];
            a3v[r] *= ar[r]; a4v[r] *= ar[r];
        }
        // ---- PV: acc(16q x 16d) += P(16q x 32k) * V(32k x 16d), 5 d-tiles (from LDS) ----
        const _Float16* vb = vsh + hi * 16;
        a0  = wmma32(p, *(const v16h*)(vb + ( 0 + l15) * 32), a0);
        a1v = wmma32(p, *(const v16h*)(vb + (16 + l15) * 32), a1v);
        a2v = wmma32(p, *(const v16h*)(vb + (32 + l15) * 32), a2v);
        a3v = wmma32(p, *(const v16h*)(vb + (48 + l15) * 32), a3v);
        a4v = wmma32(p, *(const v16h*)(vb + (64 + l15) * 32), a4v);
    }
    // ---- normalize by 1/l, store f16 for the proj GEMM ----
    float linv = 1.0f / l;
    float lr[8];
    #pragma unroll
    for (int r = 0; r < 8; r++) lr[r] = __shfl(linv, r + hi8, 32);
    #pragma unroll
    for (int r = 0; r < 8; r++) {
        size_t row = (size_t)(q0 + r + hi8) * DD + head * HD + l15;
        ao[row +  0] = (_Float16)(a0[r]  * lr[r]);
        ao[row + 16] = (_Float16)(a1v[r] * lr[r]);
        ao[row + 32] = (_Float16)(a2v[r] * lr[r]);
        ao[row + 48] = (_Float16)(a3v[r] * lr[r]);
        ao[row + 64] = (_Float16)(a4v[r] * lr[r]);
    }
}

extern "C" void kernel_launch(void* const* d_in, const int* in_sizes, int n_in,
                              void* d_out, int out_size, void* d_ws, size_t ws_size,
                              hipStream_t stream) {
    (void)in_sizes; (void)n_in; (void)out_size; (void)ws_size;
    const float* x     = (const float*)d_in[0];   // (2048,1,1280)
    const float* rope  = (const float*)d_in[1];   // (2048,40)
    const int*   seg   = (const int*)d_in[2];     // (2048,)
    const float* wqkv  = (const float*)d_in[3];   // (1280,3840)
    const float* bqkv  = (const float*)d_in[4];   // (3840,)
    const float* wproj = (const float*)d_in[5];   // (1280,1280)
    const float* bproj = (const float*)d_in[6];   // (1280,)
    float* out = (float*)d_out;                   // (2048,1,1280) f32

    char* ws = (char*)d_ws;
    size_t off = 0;
    auto carve = [&](size_t bytes) { size_t r = off; off += (bytes + 255) & ~(size_t)255; return r; };
    _Float16* x16    = (_Float16*)(ws + carve((size_t)TT * DD * 2));
    _Float16* wqkvT  = (_Float16*)(ws + carve((size_t)3 * DD * DD * 2));
    _Float16* wprojT = (_Float16*)(ws + carve((size_t)DD * DD * 2));
    float*    qkvf   = (float*)   (ws + carve((size_t)TT * 3 * DD * 4));
    _Float16* qh     = (_Float16*)(ws + carve((size_t)NH * TT * HDP * 2));
    _Float16* kh     = (_Float16*)(ws + carve((size_t)NH * TT * HDP * 2));
    _Float16* vt     = (_Float16*)(ws + carve((size_t)NH * HD * TT * 2));
    _Float16* ao16   = (_Float16*)(ws + carve((size_t)TT * DD * 2));

    // 1. precision conversion (+ weight transpose for contiguous WMMA B loads)
    k_cvt  <<<(TT * DD + 255) / 256, 256, 0, stream>>>(x, x16, TT * DD);
    k_cvt_t<<<(DD * 3 * DD + 255) / 256, 256, 0, stream>>>(wqkv, wqkvT, DD, 3 * DD);
    k_cvt_t<<<(DD * DD + 255) / 256, 256, 0, stream>>>(wproj, wprojT, DD, DD);

    // 2. QKV projection: (2048x1280) x (1280x3840) + bias, f32 out
    k_gemm<<<dim3(3 * DD / 256, TT / 64), 256, 0, stream>>>(x16, wqkvT, bqkv, qkvf, TT, 3 * DD, DD);

    // 3. RoPE + pack into WMMA-friendly layouts (Q scaled by 1/sqrt(80))
    k_rope_pack<<<TT * NH, HDP, 0, stream>>>(qkvf, rope, qh, kh, vt);

    // 4. segment-masked flash attention (16 heads x 16 q-groups = 256 blocks)
    k_attn<<<NH * (TT / 16) / 8, 256, 0, stream>>>(qh, kh, vt, seg, ao16);

    // 5. output projection to f32 d_out
    k_gemm<<<dim3(DD / 256, TT / 64), 256, 0, stream>>>(ao16, wprojT, bproj, out, TT, DD, DD);
}
